// SAM_69861938036952
// MI455X (gfx1250) — compile-verified, matrix operand order checked by reading, not run on hardware
//
#include <hip/hip_runtime.h>
#include <hip/hip_bf16.h>

typedef __attribute__((ext_vector_type(16))) __bf16 v16bf;
typedef __attribute__((ext_vector_type(8)))  float  v8f;
typedef __attribute__((ext_vector_type(4)))  unsigned int u32x4;

#define HD    264
#define HW    69696     /* 264*264 */
#define CIN   60
#define KREAL 540       /* 60 channels * 9 taps */
#define KPAD  544       /* 17 * 32  (global Fbf layout) */
#define KROW  552       /* LDS row stride: 276 DW -> conflict-free b128 reads */

__device__ __forceinline__ unsigned short f2bf(float f) {
    union { float f; unsigned int u; } c; c.f = f;
    unsigned int u = c.u;
    unsigned int r = u + 0x7FFFu + ((u >> 16) & 1u);
    return (unsigned short)(r >> 16);
}

// ---------------- Phase 1: adaptive avg pool to 3x3 -> gap[20][60][9] -------
__global__ void gap_kernel(const float* __restrict__ x, float* __restrict__ gap) {
    __shared__ float red[256];
    int idx = blockIdx.x;               // [0, 20*60*9)
    int b   = idx / (CIN * 9);
    int rem = idx % (CIN * 9);
    int c   = rem / 9;
    int tap = rem % 9;
    int th = tap / 3, tw = tap % 3;
    const float* base = x + (size_t)(b * CIN + c) * HW;
    float s = 0.f;
    for (int i = threadIdx.x; i < 88 * 88; i += 256) {
        int rr = i / 88, cc = i % 88;
        s += base[(th * 88 + rr) * HD + (tw * 88 + cc)];
    }
    red[threadIdx.x] = s;
    __syncthreads();
    for (int off = 128; off > 0; off >>= 1) {
        if ((int)threadIdx.x < off) red[threadIdx.x] += red[threadIdx.x + off];
        __syncthreads();
    }
    if (threadIdx.x == 0) gap[b * 540 + c * 9 + tap] = red[0] * (1.f / 7744.f);
}

// ------------- Phase 2a: dynamic filters f[g][O][I][tap] --------------------
__global__ void filt_kernel(const float* __restrict__ gap,
                            const float* __restrict__ w1, const float* __restrict__ b1,
                            const float* __restrict__ w2, const float* __restrict__ b2,
                            const float* __restrict__ w3, const float* __restrict__ b3,
                            const float* __restrict__ p1, const float* __restrict__ p2,
                            const float* __restrict__ p3, float* __restrict__ f) {
    int idx = blockIdx.x * 256 + threadIdx.x;
    if (idx >= 3 * 20 * 20 * 9) return;
    int g    = idx / 3600;
    int rem  = idx % 3600;
    int O    = rem / 180;               // gap-batch index == conv out channel
    int rem2 = rem % 180;
    int I    = rem2 / 9;                // weight row == conv in channel
    int tap  = rem2 % 9;
    const float* w  = (g == 0) ? w1 : (g == 1) ? w2 : w3;
    const float* bb = (g == 0) ? b1 : (g == 1) ? b2 : b3;
    float para = (g == 0) ? p1[0] : (g == 1) ? p2[0] : p3[0];
    float s = bb[I];
    for (int c = 0; c < CIN; ++c)
        s += gap[O * 540 + c * 9 + tap] * w[I * CIN + c];
    f[idx] = tanhf(para * s);
}

// ------------- Phase 2b: fold through w_out -> Fbf[64][544] bf16 ------------
__global__ void big_filter_kernel(const float* __restrict__ f,
                                  const float* __restrict__ w_out,
                                  unsigned short* __restrict__ Fbf) {
    int idx = blockIdx.x * 256 + threadIdx.x;
    if (idx >= 64 * KPAD) return;
    int oc = idx / KPAD;
    int k  = idx % KPAD;
    float v = 0.f;
    if (oc < 60 && k < KREAL) {
        int c = k / 9, tap = k % 9;
        int g = c / 20, I = c % 20;
        for (int o = 0; o < 20; ++o)
            v += w_out[oc * 60 + g * 20 + o] * f[((g * 20 + o) * 20 + I) * 9 + tap];
    }
    Fbf[idx] = f2bf(v);
}

// ------------- Phase 3: fused implicit-GEMM conv + pointwise ----------------
// im2col fill: pixel n = t&31, K-pairs strength-reduced, packed b32 stores.
__device__ __forceinline__ void fill_tile(unsigned short* __restrict__ sbuf,
                                          const float* __restrict__ xb,
                                          int pbase, int t) {
    const int n = t & 31;
    const int p = pbase + n;
    const int h = p / HD, w = p % HD;
    int k   = (t >> 5) * 2;            // even start in [0,14]
    int c   = k / 9;
    int tap = k % 9;
    unsigned short* dst = sbuf + n * KROW;
    #pragma unroll 4
    for (int i = 0; i < 34; ++i) {     // covers all even k in [0, 542]
        int c1 = c, tap1 = tap + 1;
        if (tap1 == 9) { tap1 = 0; ++c1; }
        float v0 = 0.f, v1 = 0.f;
        if (c < CIN) {
            int hh = h + tap / 3 - 1, ww = w + tap % 3 - 1;
            if (hh >= 0 && hh < HD && ww >= 0 && ww < HD)
                v0 = xb[(size_t)c * HW + hh * HD + ww];
        }
        if (c1 < CIN) {
            int hh = h + tap1 / 3 - 1, ww = w + tap1 % 3 - 1;
            if (hh >= 0 && hh < HD && ww >= 0 && ww < HD)
                v1 = xb[(size_t)c1 * HW + hh * HD + ww];
        }
        *(unsigned int*)(dst + k) =
            (unsigned int)f2bf(v0) | ((unsigned int)f2bf(v1) << 16);
        k += 16;                        // k += 16  =>  c += 1, tap += 7 (wrap)
        tap += 7; ++c;
        if (tap >= 9) { tap -= 9; ++c; }
    }
}

__global__ void __launch_bounds__(256) dynconv_kernel(
        const float* __restrict__ x,
        const unsigned short* __restrict__ Fbf,
        const float* __restrict__ b_out,
        float* __restrict__ out) {
    __shared__ __align__(16) unsigned short sB[2][32 * KROW];   // 70656 B

    const int b    = blockIdx.y;
    const int t    = threadIdx.x;
    const int wave = t >> 5;
    const int lane = t & 31;
    const int half = lane >> 4;
    const int r    = lane & 15;
    const int mtile = wave & 3;         // 16 of 64 output channels
    const int ntile = wave >> 2;        // 16 of 32 pixels
    const float* xb = x + (size_t)b * CIN * HW;
    const int tile0 = blockIdx.x * 6;

    union Frag { struct { u32x4 lo, hi; } q; v16bf v; };

    fill_tile(sB[0], xb, tile0 * 32, t);

    for (int tl = 0; tl < 6; ++tl) {
        __syncthreads();                      // fence prior reads + fills
        if (tl < 5)                           // produce next tile's operand
            fill_tile(sB[(tl + 1) & 1], xb, (tile0 + tl + 1) * 32, t);

        // ---- consume current tile: 17 WMMA K-steps, B double-buffered ----
        const unsigned short* sBrow = sB[tl & 1] + (ntile * 16 + r) * KROW;
        const unsigned short* Arow  = Fbf + (mtile * 16 + r) * KPAD;

        Frag fb[2];
        fb[0].q.lo = *(const u32x4*)(sBrow + half * 8);
        fb[0].q.hi = *(const u32x4*)(sBrow + 16 + half * 8);
        v8f acc = {};
        #pragma unroll
        for (int kt = 0; kt < 17; ++kt) {
            const int kb  = kt * 32;
            const int cur = kt & 1, nxt = cur ^ 1;
            if (kt < 16) {                    // prefetch next B fragment
                fb[nxt].q.lo = *(const u32x4*)(sBrow + kb + 32 + half * 8);
                fb[nxt].q.hi = *(const u32x4*)(sBrow + kb + 48 + half * 8);
            }
            Frag fa;                          // A hoisted to VGPRs by unroll
            fa.q.lo = *(const u32x4*)(Arow + kb + half * 8);
            fa.q.hi = *(const u32x4*)(Arow + kb + 16 + half * 8);
            acc = __builtin_amdgcn_wmma_f32_16x16x32_bf16(
                false, fa.v, false, fb[cur].v, (short)0, acc, false, false);
        }

        // D layout: VGPR e, lanes0-15 -> M=e, lanes16-31 -> M=e+8; N=lane&15
        const int p = (tile0 + tl) * 32 + ntile * 16 + r;
        #pragma unroll
        for (int e = 0; e < 8; ++e) {
            int oc = mtile * 16 + e + 8 * half;
            if (oc < 60)
                out[((size_t)b * 60 + oc) * HW + p] = acc[e] + b_out[oc];
        }
    }
}

extern "C" void kernel_launch(void* const* d_in, const int* in_sizes, int n_in,
                              void* d_out, int out_size, void* d_ws, size_t ws_size,
                              hipStream_t stream) {
    const float* x     = (const float*)d_in[0];
    const float* w1    = (const float*)d_in[1];
    const float* b1    = (const float*)d_in[2];
    const float* w2    = (const float*)d_in[3];
    const float* b2    = (const float*)d_in[4];
    const float* w3    = (const float*)d_in[5];
    const float* b3    = (const float*)d_in[6];
    const float* p1    = (const float*)d_in[7];
    const float* p2    = (const float*)d_in[8];
    const float* p3    = (const float*)d_in[9];
    const float* w_out = (const float*)d_in[10];
    const float* b_out = (const float*)d_in[11];
    float* out = (float*)d_out;

    char* ws = (char*)d_ws;
    float* gap          = (float*)ws;                    // 10800 f = 43200 B
    float* f            = (float*)(ws + 43200);          // 10800 f = 43200 B
    unsigned short* Fbf = (unsigned short*)(ws + 86400); // 34816 bf16 = 69632 B

    gap_kernel<<<20 * 60 * 9, 256, 0, stream>>>(x, gap);
    filt_kernel<<<(10800 + 255) / 256, 256, 0, stream>>>(
        gap, w1, b1, w2, b2, w3, b3, p1, p2, p3, f);
    big_filter_kernel<<<(64 * KPAD + 255) / 256, 256, 0, stream>>>(f, w_out, Fbf);

    dim3 grid(363, 20);   // 363*6 = 2178 tiles of 32 pixels = 69696; y = batch
    dynconv_kernel<<<grid, 256, 0, stream>>>(x, Fbf, b_out, out);
}
